// Mamba_24266565223039
// MI455X (gfx1250) — matrix operand order repaired and checked by reading, not compiled, not run on hardware
//
#include <hip/hip_runtime.h>
#include <hip/hip_bf16.h>

// ---------------------------------------------------------------------------
// Bidirectional Mamba block for gfx1250 (MI455X).
//   GEMMs  : v_wmma_f32_16x16x32_bf16, weights pre-converted to bf16 in ws,
//            activations packed f32->bf16 via v_perm_b32 (1 op / 2 elems)
//   Scan   : 1 lane per (b, channel, state); shfl_xor butterfly over N=16
//   Layouts: token-major so K is contiguous for every GEMM operand
// ---------------------------------------------------------------------------

typedef __attribute__((ext_vector_type(16))) __bf16 v16bf;
typedef __attribute__((ext_vector_type(8)))  __bf16 v8bf;
typedef __attribute__((ext_vector_type(8)))  float  v8f;
typedef __attribute__((ext_vector_type(4)))  float  v4f;

union V16U { v16bf v; v8bf h[2]; unsigned u[8]; };

constexpr int BSZ  = 2;     // batch
constexpr int LSEQ = 2048;  // sequence length
constexpr int DM   = 1024;  // d_model
constexpr int DI   = 2048;  // d_inner
constexpr int NST  = 16;    // d_state
constexpr int RNK  = 64;    // dt_rank
constexpr int KC   = 4;     // conv kernel

// pack two f32 into packed bf16 pair (truncation) with a single v_perm_b32:
// result = { hi[31:16], lo[31:16] }
__device__ __forceinline__ unsigned pack2bf(float lo, float hi) {
  return __builtin_amdgcn_perm(__builtin_bit_cast(unsigned, hi),
                               __builtin_bit_cast(unsigned, lo),
                               0x07060302u);
}

__device__ __forceinline__ float softplusf(float x) {
  return (x > 20.f) ? x : __logf(1.f + __expf(x));
}
__device__ __forceinline__ float siluf(float x) {
  return x / (1.f + __expf(-x));
}

// ---------------------------------------------------------------------------
// One-shot f32 -> bf16 weight conversion (2 elements / thread via v_perm).
// count must be even (true for all weight matrices here).
// ---------------------------------------------------------------------------
__global__ __launch_bounds__(256)
void w_to_bf16(const float* __restrict__ in, unsigned* __restrict__ out, int pairs)
{
  const int gid = blockIdx.x * 256 + threadIdx.x;
  if (gid < pairs) {
    const float lo = in[2 * gid];
    const float hi = in[2 * gid + 1];
    out[gid] = pack2bf(lo, hi);
  }
}

// ---------------------------------------------------------------------------
// A-fragment builder: 16x32 bf16 fragment from f32 row-major A.
// ISA layout (CDNA5 ISA §7.12.2): lane<16 -> K in {0..7,16..23}+ ,
// lane>=16 -> +8; elements 0..7 = first K-octet, 8..15 = second.
// ---------------------------------------------------------------------------
__device__ __forceinline__ v16bf make_afrag(const float* ap) {
  const v4f a0 = *(const v4f*)(ap);
  const v4f a1 = *(const v4f*)(ap + 4);
  const v4f a2 = *(const v4f*)(ap + 16);
  const v4f a3 = *(const v4f*)(ap + 20);
  V16U u;
  u.u[0] = pack2bf(a0[0], a0[1]);
  u.u[1] = pack2bf(a0[2], a0[3]);
  u.u[2] = pack2bf(a1[0], a1[1]);
  u.u[3] = pack2bf(a1[2], a1[3]);
  u.u[4] = pack2bf(a2[0], a2[1]);
  u.u[5] = pack2bf(a2[2], a2[3]);
  u.u[6] = pack2bf(a3[0], a3[1]);
  u.u[7] = pack2bf(a3[2], a3[3]);
  return u.v;
}

// B-fragment: two 8-bf16 octets (K 0..7 and 16..23 from wp) straight from converted weights.
__device__ __forceinline__ v16bf make_bfrag(const __bf16* wp) {
  V16U u;
  u.h[0] = *(const v8bf*)(wp);
  u.h[1] = *(const v8bf*)(wp + 16);
  return u.v;
}

// ---------------------------------------------------------------------------
// Generic GEMM:  C[b][m][n] = sum_k A[b][m][k] * Wbf[n][k]
// Block = 256 threads = 8 waves; block tile 64(M) x 128(N);
// wave tile 16(M) x 64(N) = 4 accumulators, K step 32.
// M multiple of 64 (all uses: M == LSEQ). FULLN=false adds N-tail guards.
// ---------------------------------------------------------------------------
template <bool FULLN>
__global__ __launch_bounds__(256)
void gemm_awt(const float* __restrict__ A, int lda, long long strideA,
              const __bf16* __restrict__ W, int ldw,
              float* __restrict__ C, int ldc, long long strideC,
              int N, int K)
{
  const int tid  = threadIdx.x;
  const int lane = tid & 31;
  const int wave = tid >> 5;
  const int wm   = wave & 3;   // 4 waves along M
  const int wn   = wave >> 2;  // 2 waves along N

  A += (long long)blockIdx.z * strideA;
  C += (long long)blockIdx.z * strideC;

  const int tileM  = blockIdx.y * 64 + wm * 16;
  const int tileN0 = blockIdx.x * 128 + wn * 64;
  const int mrow   = tileM + (lane & 15);
  const int aOff   = (lane & 16) ? 8 : 0;    // A K-half select per ISA layout
  const int bOff   = (lane & 16) ? 8 : 0;    // B K-half select per ISA layout

  const float* arow = A + (long long)mrow * lda + aOff;

  v8f acc[4] = {};

  for (int k0 = 0; k0 < K; k0 += 32) {
    if (k0 + 32 < K) __builtin_prefetch(arow + k0 + 32, 0, 3);
    const v16bf af = make_afrag(arow + k0);
#pragma unroll
    for (int t = 0; t < 4; ++t) {
      const int ncol = tileN0 + t * 16 + (lane & 15);
      v16bf bf;
      if (FULLN || ncol < N) {
        bf = make_bfrag(W + (long long)ncol * ldw + k0 + bOff);
      } else {
        V16U z = {};
        bf = z.v;
      }
      acc[t] = __builtin_amdgcn_wmma_f32_16x16x32_bf16(
          false, af, false, bf, (short)0, acc[t], false, false);
    }
  }

  const int mbase = tileM + ((lane & 16) ? 8 : 0);
#pragma unroll
  for (int t = 0; t < 4; ++t) {
    const int ncol = tileN0 + t * 16 + (lane & 15);
    if (!FULLN && ncol >= N) continue;
#pragma unroll
    for (int r = 0; r < 8; ++r)
      C[(long long)(mbase + r) * ldc + ncol] = acc[t][r];
  }
}

// ---------------------------------------------------------------------------
// Fused out_proj: C[b][l][m] = sum_e (yf[b][l][e] + yr[b][L-1-l][e]) * W[m][e]
// Branch-sum and backward un-flip folded into the A load (saves a 64 MB
// HBM round trip). N = DM (full tiles), K = DI.
// ---------------------------------------------------------------------------
__global__ __launch_bounds__(256)
void gemm_out(const float* __restrict__ Yf, const float* __restrict__ Yr,
              const __bf16* __restrict__ W, float* __restrict__ C)
{
  const int tid  = threadIdx.x;
  const int lane = tid & 31;
  const int wave = tid >> 5;
  const int wm   = wave & 3;
  const int wn   = wave >> 2;

  const long long bo = (long long)blockIdx.z * LSEQ * DI;
  const int tileM  = blockIdx.y * 64 + wm * 16;
  const int tileN0 = blockIdx.x * 128 + wn * 64;
  const int mrow   = tileM + (lane & 15);
  const int aOff   = (lane & 16) ? 8 : 0;
  const int bOff   = (lane & 16) ? 8 : 0;

  const float* apf = Yf + bo + (long long)mrow * DI + aOff;
  const float* apr = Yr + bo + (long long)(LSEQ - 1 - mrow) * DI + aOff;

  v8f acc[4] = {};

  for (int k0 = 0; k0 < DI; k0 += 32) {
    const v4f f0 = *(const v4f*)(apf + k0);
    const v4f f1 = *(const v4f*)(apf + k0 + 4);
    const v4f f2 = *(const v4f*)(apf + k0 + 16);
    const v4f f3 = *(const v4f*)(apf + k0 + 20);
    const v4f r0 = *(const v4f*)(apr + k0);
    const v4f r1 = *(const v4f*)(apr + k0 + 4);
    const v4f r2 = *(const v4f*)(apr + k0 + 16);
    const v4f r3 = *(const v4f*)(apr + k0 + 20);
    V16U au;
    au.u[0] = pack2bf(f0[0] + r0[0], f0[1] + r0[1]);
    au.u[1] = pack2bf(f0[2] + r0[2], f0[3] + r0[3]);
    au.u[2] = pack2bf(f1[0] + r1[0], f1[1] + r1[1]);
    au.u[3] = pack2bf(f1[2] + r1[2], f1[3] + r1[3]);
    au.u[4] = pack2bf(f2[0] + r2[0], f2[1] + r2[1]);
    au.u[5] = pack2bf(f2[2] + r2[2], f2[3] + r2[3]);
    au.u[6] = pack2bf(f3[0] + r3[0], f3[1] + r3[1]);
    au.u[7] = pack2bf(f3[2] + r3[2], f3[3] + r3[3]);
    const v16bf af = au.v;
#pragma unroll
    for (int t = 0; t < 4; ++t) {
      const int ncol = tileN0 + t * 16 + (lane & 15);
      const v16bf bf = make_bfrag(W + (long long)ncol * DI + k0 + bOff);
      acc[t] = __builtin_amdgcn_wmma_f32_16x16x32_bf16(
          false, af, false, bf, (short)0, acc[t], false, false);
    }
  }

  const long long co = (long long)blockIdx.z * LSEQ * DM;
  const int mbase = tileM + ((lane & 16) ? 8 : 0);
#pragma unroll
  for (int t = 0; t < 4; ++t) {
    const int ncol = tileN0 + t * 16 + (lane & 15);
#pragma unroll
    for (int r = 0; r < 8; ++r)
      C[co + (long long)(mbase + r) * DM + ncol] = acc[t][r];
  }
}

// ---------------------------------------------------------------------------
// Depthwise causal conv (K=4) + SiLU, in branch coordinates (flip=1 reads
// the sequence reversed; downstream branch buffers live in flipped space).
// ---------------------------------------------------------------------------
__global__ __launch_bounds__(256)
void conv_silu(const float* __restrict__ xz, const float* __restrict__ cw,
               const float* __restrict__ cb, float* __restrict__ xc, int flip)
{
  const long long gid = (long long)blockIdx.x * 256 + threadIdx.x; // (b,l,e)
  const int e = (int)(gid % DI);
  const long long bl = gid / DI;
  const int l = (int)(bl % LSEQ);
  const int b = (int)(bl / LSEQ);

  float acc = cb[e];
#pragma unroll
  for (int t = 0; t < KC; ++t) {
    const int li = l - (KC - 1) + t;
    if (li >= 0) {
      const int lsrc = flip ? (LSEQ - 1 - li) : li;
      acc += cw[e * KC + t] * xz[((long long)b * LSEQ + lsrc) * (2 * DI) + e];
    }
  }
  xc[gid] = siluf(acc);
}

// ---------------------------------------------------------------------------
// Selective scan. One lane per (b, channel e, state n): 65536 lanes =
// 2048 wave32s. Per step: one v_exp (decay) + few FMAs per lane; dot over
// N=16 via 4-step shfl_xor butterfly. Lane n==0 applies D-skip + silu(z)
// gating and stores.
// ---------------------------------------------------------------------------
__global__ __launch_bounds__(256)
void scan_kernel(const float* __restrict__ dpre, const float* __restrict__ dtb,
                 const float* __restrict__ xconv, const float* __restrict__ xdbl,
                 const float* __restrict__ xz, const float* __restrict__ A_log,
                 const float* __restrict__ Dp, float* __restrict__ y, int flip)
{
  const int tid = threadIdx.x;
  const int n   = tid & 15;                       // state index
  const int e   = blockIdx.x * 16 + (tid >> 4);   // channel
  const int b   = blockIdx.y;

  const float Am     = -__expf(A_log[e * NST + n]);
  const float dtbias = dtb[e];
  const float dpe    = Dp[e];
  float h = 0.f;

  const long long rowb = (long long)b * LSEQ;
  for (int l = 0; l < LSEQ; ++l) {
    const long long idx = rowb + l;
    const float d  = softplusf(dpre[idx * DI + e] + dtbias);
    const float u  = xconv[idx * DI + e];
    const float bn = xdbl[idx * 96 + RNK + n];        // B_t[n]
    const float cn = xdbl[idx * 96 + RNK + NST + n];  // C_t[n]

    h = h * __expf(d * Am) + (d * u) * bn;

    float p = h * cn;
    p += __shfl_xor(p, 1, 16);
    p += __shfl_xor(p, 2, 16);
    p += __shfl_xor(p, 4, 16);
    p += __shfl_xor(p, 8, 16);

    if (n == 0) {
      const long long zrow = flip ? (rowb + (LSEQ - 1 - l)) : idx;
      const float z = xz[zrow * (2 * DI) + DI + e];
      y[idx * DI + e] = (p + u * dpe) * siluf(z);
    }
  }
}

// ---------------------------------------------------------------------------
extern "C" void kernel_launch(void* const* d_in, const int* in_sizes, int n_in,
                              void* d_out, int out_size, void* d_ws, size_t ws_size,
                              hipStream_t stream) {
  (void)in_sizes; (void)n_in; (void)out_size; (void)ws_size;

  const float* hs        = (const float*)d_in[0];
  const float* in_proj_w = (const float*)d_in[1];
  const float* conv_w_f  = (const float*)d_in[2];
  const float* conv_b_f  = (const float*)d_in[3];
  const float* x_proj_f  = (const float*)d_in[4];
  const float* dt_w_f    = (const float*)d_in[5];
  const float* dt_b_f    = (const float*)d_in[6];
  const float* A_log_f   = (const float*)d_in[7];
  const float* D_f       = (const float*)d_in[8];
  const float* conv_w_r  = (const float*)d_in[9];
  const float* conv_b_r  = (const float*)d_in[10];
  const float* x_proj_r  = (const float*)d_in[11];
  const float* dt_w_r    = (const float*)d_in[12];
  const float* dt_b_r    = (const float*)d_in[13];
  const float* A_log_r   = (const float*)d_in[14];
  const float* D_r       = (const float*)d_in[15];
  const float* out_w     = (const float*)d_in[16];
  float* out = (float*)d_out;

  // ---- workspace layout ----
  float* ws    = (float*)d_ws;
  float* xz    = ws;                                   // B*L*2*DI
  float* xconv = xz    + (size_t)BSZ * LSEQ * 2 * DI;  // B*L*DI
  float* xdbl  = xconv + (size_t)BSZ * LSEQ * DI;      // B*L*96
  float* dpre  = xdbl  + (size_t)BSZ * LSEQ * 96;      // B*L*DI
  float* yf    = dpre  + (size_t)BSZ * LSEQ * DI;      // B*L*DI
  float* yr    = yf    + (size_t)BSZ * LSEQ * DI;      // B*L*DI
  __bf16* wbf  = (__bf16*)(yr + (size_t)BSZ * LSEQ * DI);

  constexpr size_t N_IN  = (size_t)(2 * DI) * DM;  // in_proj
  constexpr size_t N_XP  = (size_t)96 * DI;        // x_proj
  constexpr size_t N_DT  = (size_t)DI * RNK;       // dt_proj
  constexpr size_t N_OUT = (size_t)DM * DI;        // out_proj

  __bf16* wbf_in   = wbf;
  __bf16* wbf_xp_f = wbf_in   + N_IN;
  __bf16* wbf_xp_r = wbf_xp_f + N_XP;
  __bf16* wbf_dt_f = wbf_xp_r + N_XP;
  __bf16* wbf_dt_r = wbf_dt_f + N_DT;
  __bf16* wbf_out  = wbf_dt_r + N_DT;

  const dim3 blk(256);
  auto cvt = [&](const float* src, __bf16* dst, size_t count) {
    const int pairs = (int)(count / 2);
    w_to_bf16<<<(pairs + 255) / 256, blk, 0, stream>>>(src, (unsigned*)dst, pairs);
  };
  cvt(in_proj_w, wbf_in,   N_IN);
  cvt(x_proj_f,  wbf_xp_f, N_XP);
  cvt(x_proj_r,  wbf_xp_r, N_XP);
  cvt(dt_w_f,    wbf_dt_f, N_DT);
  cvt(dt_w_r,    wbf_dt_r, N_DT);
  cvt(out_w,     wbf_out,  N_OUT);

  // in_proj: xz[b][l][e] = sum_d hs[b][l][d] * in_proj_w[e][d]
  gemm_awt<true><<<dim3((2 * DI) / 128, LSEQ / 64, BSZ), blk, 0, stream>>>(
      hs, DM, (long long)LSEQ * DM,
      wbf_in, DM,
      xz, 2 * DI, (long long)LSEQ * 2 * DI,
      2 * DI, DM);

  for (int br = 0; br < 2; ++br) {
    const float*  cw   = br ? conv_w_r : conv_w_f;
    const float*  cbv  = br ? conv_b_r : conv_b_f;
    const __bf16* xpw  = br ? wbf_xp_r : wbf_xp_f;
    const __bf16* dtw  = br ? wbf_dt_r : wbf_dt_f;
    const float*  dtbv = br ? dt_b_r   : dt_b_f;
    const float*  alg  = br ? A_log_r  : A_log_f;
    const float*  dvc  = br ? D_r      : D_f;
    float* ybuf        = br ? yr       : yf;

    // causal depthwise conv + silu (branch coordinates)
    conv_silu<<<(BSZ * LSEQ * DI) / 256, blk, 0, stream>>>(xz, cw, cbv, xconv, br);

    // x_proj: xdbl[b][l][r] = sum_d xconv[b][l][d] * x_proj_w[r][d]  (N=96 tail)
    gemm_awt<false><<<dim3(1, LSEQ / 64, BSZ), blk, 0, stream>>>(
        xconv, DI, (long long)LSEQ * DI,
        xpw, DI,
        xdbl, 96, (long long)LSEQ * 96,
        96, DI);

    // dt_proj: dpre[b][l][e] = sum_r xdbl[b][l][r<64] * dt_proj_w[e][r]
    gemm_awt<true><<<dim3(DI / 128, LSEQ / 64, BSZ), blk, 0, stream>>>(
        xdbl, 96, (long long)LSEQ * 96,
        dtw, RNK,
        dpre, DI, (long long)LSEQ * DI,
        DI, RNK);

    // selective scan + D skip + silu(z) gating
    scan_kernel<<<dim3(DI / 16, BSZ), blk, 0, stream>>>(
        dpre, dtbv, xconv, xdbl, xz, alg, dvc, ybuf, br);
  }

  // out_proj with fused (y_f + flip(y_r)) in the A operand
  gemm_out<<<dim3(DM / 128, LSEQ / 64, BSZ), blk, 0, stream>>>(yf, yr, wbf_out, out);
}
